// Block_67577015435503
// MI455X (gfx1250) — compile-verified, matrix operand order checked by reading, not run on hardware
//
#include <hip/hip_runtime.h>
#include <hip/hip_bf16.h>

typedef __attribute__((ext_vector_type(16))) __bf16 v16bf;
typedef __attribute__((ext_vector_type(8)))  __bf16 v8bf;
typedef __attribute__((ext_vector_type(8)))  float  v8f;
typedef __attribute__((ext_vector_type(4)))  unsigned int v4u;
typedef __attribute__((ext_vector_type(8)))  int v8i;
typedef __attribute__((ext_vector_type(4)))  int v4i;

#if defined(__has_builtin)
#if __has_builtin(__builtin_amdgcn_tensor_load_to_lds)
#define ATHENA_HAVE_TDM 1
#else
#define ATHENA_HAVE_TDM 0
#endif
#else
#define ATHENA_HAVE_TDM 0
#endif

// ---------------------------------------------------------------------------
// CDNA5 wave32 WMMA: D[16x16,f32] = A[16x32,bf16] * B[32x16,bf16] + C
// A (16x32 MxK): lane m / 16+m hold row m; lanes 0-15 -> K {0..7,16..23},
//                lanes 16-31 -> K {8..15,24..31}  (ISA 7.12.2)
// B (32x16 KxN): lane holds col n = lane&15; K chunks split by lane half.
// C/D: VGPR i -> row M = i + (lane>=16 ? 8 : 0), N = lane&15.
// ---------------------------------------------------------------------------

__device__ inline v8f wmma_bf16(v16bf a, v16bf b, v8f c) {
  return __builtin_amdgcn_wmma_f32_16x16x32_bf16(
      false, a, false, b, (short)0, c, false, false);
}

// A-fragment from row-major [16 x ld] bf16 tile: two 16B ds_load_b128 per lane.
__device__ inline v16bf frag_rowK(const __bf16* __restrict__ p, int ld) {
  const int lane = threadIdx.x & 31;
  const int r  = lane & 15;
  const int k0 = (lane & 16) ? 8 : 0;
  const __bf16* q = p + r * ld + k0;
  v8bf lo = *(const v8bf*)(q);
  v8bf hi = *(const v8bf*)(q + 16);
  return __builtin_shufflevector(lo, hi, 0, 1, 2, 3, 4, 5, 6, 7,
                                 8, 9, 10, 11, 12, 13, 14, 15);
}

// A-fragment from fp32 LDS (softmax probs), converted via v_cvt_pk_bf16_f32.
__device__ inline v16bf frag_rowK_f32(const float* __restrict__ p, int ld) {
  const int lane = threadIdx.x & 31;
  const int r  = lane & 15;
  const int k0 = (lane & 16) ? 8 : 0;
  const float* q = p + r * ld + k0;
  v16bf f;
#pragma unroll
  for (int i = 0; i < 8; ++i) f[i]     = (__bf16)q[i];
#pragma unroll
  for (int i = 0; i < 8; ++i) f[i + 8] = (__bf16)q[16 + i];
  return f;
}

// Swizzled (fragment-major) B storage: 32(K) x 64(N) tile -> for n-subtile nt
// and lane L, the 16 fragment elements live contiguously at (nt*32+L)*16.
__device__ inline int bswz(int k, int n) {     // k in [0,32), n in [0,64)
  const int nt   = n >> 4;
  const int half = (k >> 3) & 1;               // which lane half wants this k
  const int lane = (half << 4) | (n & 15);
  const int idx  = ((k >> 4) << 3) | (k & 7);  // element index within fragment
  return ((nt << 5) | lane) * 16 + idx;
}

// B-fragment read from swizzled tile: two contiguous 16B ds_load_b128.
__device__ inline v16bf frag_swz(const __bf16* __restrict__ base, int nt) {
  const int lane = threadIdx.x & 31;
  const __bf16* q = base + (((nt << 5) | lane) << 4);
  v8bf lo = *(const v8bf*)(q);
  v8bf hi = *(const v8bf*)(q + 8);
  return __builtin_shufflevector(lo, hi, 0, 1, 2, 3, 4, 5, 6, 7,
                                 8, 9, 10, 11, 12, 13, 14, 15);
}

#if ATHENA_HAVE_TDM
// ---------------------------------------------------------------------------
// Tensor Data Mover: DMA a 2D bf16 tile (tile_w x tile_h, row stride
// `stride_elems`) from global memory into LDS at byte offset `lds_off`.
// Descriptor per CDNA5 ISA 8.3/8.4 (D# group0 + group1, groups 2/3 unused).
// Issue from one wave; completion via s_wait_tensorcnt.
// ---------------------------------------------------------------------------
__device__ inline void tdm_load_tile_bf16(const void* gptr, unsigned lds_off,
                                          unsigned tile_w, unsigned tile_h,
                                          unsigned long long stride_elems,
                                          unsigned tensor_w, unsigned tensor_h) {
  const unsigned long long ga = (unsigned long long)gptr;
  v4u g0;
  g0[0] = 1u;                                            // count=1 (valid D#)
  g0[1] = lds_off;                                       // lds_addr (bytes)
  g0[2] = (unsigned)(ga & 0xffffffffull);                // global_addr[31:0]
  g0[3] = (unsigned)((ga >> 32) & 0x01fffffful)          // global_addr[56:32]
          | (2u << 30);                                  // type = 2 ("image")
  v8i g1;
  g1[0] = (int)(1u << 16);                               // data_size=1 -> 2B
  g1[1] = (int)((tensor_w & 0xffffu) << 16);             // tensor_dim0[15:0]
  g1[2] = (int)((tensor_w >> 16) | ((tensor_h & 0xffffu) << 16));
  g1[3] = (int)((tensor_h >> 16) | (tile_w << 16));      // tile_dim0
  g1[4] = (int)(tile_h);                                 // tile_dim1 (dim2=0)
  g1[5] = (int)(unsigned)(stride_elems & 0xffffffffull); // dim0_stride[31:0]
  g1[6] = (int)(unsigned)((stride_elems >> 32) & 0xffffull);
  g1[7] = 0;
  v4i z4 = {};
#if defined(__clang_major__) && (__clang_major__ >= 23)
  v8i z8 = {};
  __builtin_amdgcn_tensor_load_to_lds(g0, g1, z4, z4, z8, 0);
#else
  __builtin_amdgcn_tensor_load_to_lds(g0, g1, z4, z4, 0);
#endif
}
#endif

// ---------------------------------------------------------------------------
// fp32 -> bf16 conversion (one-time weight conversion)
// ---------------------------------------------------------------------------
__global__ void f32_to_bf16(const float* __restrict__ in,
                            __bf16* __restrict__ out, int n) {
  int i = blockIdx.x * blockDim.x + threadIdx.x;
  if (i < n) out[i] = (__bf16)in[i];
}

// ---------------------------------------------------------------------------
// LayerNorm over C=1024, one block per row, bf16 output
// ---------------------------------------------------------------------------
__global__ __launch_bounds__(256) void layernorm_bf16(
    const float* __restrict__ x, const float* __restrict__ g,
    const float* __restrict__ bta, __bf16* __restrict__ out) {
  __shared__ float s1[256];
  __shared__ float s2[256];
  const int tid = threadIdx.x;
  const size_t row = blockIdx.x;
  const float* xr = x + row * 1024;
  float sum = 0.f, sq = 0.f;
#pragma unroll
  for (int j = 0; j < 4; ++j) {
    float v = xr[tid + j * 256];
    sum += v; sq += v * v;
  }
  s1[tid] = sum; s2[tid] = sq;
  __syncthreads();
  for (int s = 128; s > 0; s >>= 1) {
    if (tid < s) { s1[tid] += s1[tid + s]; s2[tid] += s2[tid + s]; }
    __syncthreads();
  }
  const float mu   = s1[0] * (1.f / 1024.f);
  const float var  = s2[0] * (1.f / 1024.f) - mu * mu;
  const float rstd = rsqrtf(var + 1e-5f);
#pragma unroll
  for (int j = 0; j < 4; ++j) {
    int idx = tid + j * 256;
    out[row * 1024 + idx] = (__bf16)((xr[idx] - mu) * rstd * g[idx] + bta[idx]);
  }
}

// ---------------------------------------------------------------------------
// Tiled WMMA GEMM: out[M,N] = act(A[M,K] @ Bw[K,N] + bias (+ resid))
// Block = 256 threads = 8 waves; block tile 128(M) x 64(N); K-step 32.
// Wave w: M-subtiles (w&3)*32 + {0,16}; N-subtiles (w>>2)*32 + {0,16}
// -> 4 accumulators, 4 wmma per K-step per wave.
// A tile staged via Tensor Data Mover (async, TENSORcnt) when available,
// overlapped with the manual swizzle-scatter staging of the B tile.
// ---------------------------------------------------------------------------
template <bool RELU, bool RESID>
__global__ __launch_bounds__(256) void gemm_wmma(
    const __bf16* __restrict__ A, const __bf16* __restrict__ Bw,
    const float* __restrict__ bias, const float* __restrict__ resid,
    float* __restrict__ outF, __bf16* __restrict__ outB,
    int M, int N, int K) {
  __shared__ __bf16 As[128 * 32];  // row-major
  __shared__ __bf16 Bs[32 * 64];   // swizzled fragment-major
  const int tid = threadIdx.x;
  const int w   = tid >> 5;
  const int m0  = blockIdx.y * 128;
  const int n0  = blockIdx.x * 64;
  const int mt  = (w & 3) * 32;
  const int nb  = (w >> 2) * 32;
  const int nt0 = nb >> 4;
  v8f c00 = {}, c01 = {}, c10 = {}, c11 = {};

  // staging coordinates
  const int ar = tid >> 1, ac = (tid & 1) * 16;   // A: 16 elems/thread
  const int bk = tid >> 3, bn = (tid & 7) * 8;    // B: 8 elems/thread
#if ATHENA_HAVE_TDM
  const unsigned asOff = (unsigned)(uintptr_t)(void*)&As[0];
#endif

  for (int k0 = 0; k0 < K; k0 += 32) {
#if ATHENA_HAVE_TDM
    if (w == 0) {  // async DMA of the 128x32 A tile, row stride K
      tdm_load_tile_bf16(A + (size_t)m0 * K + k0, asOff,
                         /*tile_w=*/32, /*tile_h=*/128,
                         /*stride=*/(unsigned long long)K,
                         /*tensor_w=*/32, /*tensor_h=*/128);
    }
#else
    {  // A tile 128x32, vectorized 16B copies
      const v8bf* g = (const v8bf*)(A + (size_t)(m0 + ar) * K + k0 + ac);
      *(v8bf*)(As + ar * 32 + ac)     = g[0];
      *(v8bf*)(As + ar * 32 + ac + 8) = g[1];
    }
#endif
    {  // B tile 32x64, vector load + swizzled scatter
      v8bf v = *(const v8bf*)(Bw + (size_t)(k0 + bk) * N + n0 + bn);
#pragma unroll
      for (int j = 0; j < 8; ++j) Bs[bswz(bk, bn + j)] = v[j];
    }
    if (k0 + 32 < K) {  // global_prefetch_b8 for next tiles
      __builtin_prefetch(&A[(size_t)(m0 + ar) * K + k0 + 32 + ac], 0, 1);
      __builtin_prefetch(&Bw[(size_t)(k0 + 32 + bk) * N + n0 + bn], 0, 1);
    }
#if ATHENA_HAVE_TDM
    if (w == 0) __builtin_amdgcn_s_wait_tensorcnt((short)0);
#endif
    __syncthreads();
    v16bf a0 = frag_rowK(As + mt * 32, 32);
    v16bf a1 = frag_rowK(As + (mt + 16) * 32, 32);
    v16bf b0 = frag_swz(Bs, nt0);
    v16bf b1 = frag_swz(Bs, nt0 + 1);
    c00 = wmma_bf16(a0, b0, c00);
    c01 = wmma_bf16(a0, b1, c01);
    c10 = wmma_bf16(a1, b0, c10);
    c11 = wmma_bf16(a1, b1, c11);
    __syncthreads();
  }

  const int lane   = tid & 31;
  const int m_half = (lane & 16) ? 8 : 0;
  const int n      = lane & 15;
  const int cA = n0 + nb + n;
  const int cB = cA + 16;
  const float biasA = bias[cA], biasB = bias[cB];
#pragma unroll
  for (int i = 0; i < 8; ++i) {
    const int rA = m0 + mt + i + m_half;
    const int rB = rA + 16;
    float v00 = c00[i] + biasA, v01 = c01[i] + biasB;
    float v10 = c10[i] + biasA, v11 = c11[i] + biasB;
    if (RELU) {
      v00 = fmaxf(v00, 0.f); v01 = fmaxf(v01, 0.f);
      v10 = fmaxf(v10, 0.f); v11 = fmaxf(v11, 0.f);
    }
    const size_t iA0 = (size_t)rA * N + cA, iA1 = (size_t)rA * N + cB;
    const size_t iB0 = (size_t)rB * N + cA, iB1 = (size_t)rB * N + cB;
    if (RESID) {
      outF[iA0] = v00 + resid[iA0]; outF[iA1] = v01 + resid[iA1];
      outF[iB0] = v10 + resid[iB0]; outF[iB1] = v11 + resid[iB1];
    } else {
      outB[iA0] = (__bf16)v00; outB[iA1] = (__bf16)v01;
      outB[iB0] = (__bf16)v10; outB[iB1] = (__bf16)v11;
    }
  }
}

// ---------------------------------------------------------------------------
// Flash attention: one block per (q-tile of 64, b*H). qkv bf16 [B,T,3C].
// Online softmax, causal. QK^T and PV via v_wmma_f32_16x16x32_bf16.
// Vs is stored swizzled (two 32-row sub-tiles) for contiguous B-fragments.
// ---------------------------------------------------------------------------
__global__ __launch_bounds__(256) void flash_attn(
    const __bf16* __restrict__ qkv, __bf16* __restrict__ y) {
  __shared__ __bf16 Qs[64 * 64];   // row-major [q][d]
  __shared__ __bf16 Ks[64 * 64];   // row-major [kv][d]
  __shared__ __bf16 Vs[64 * 64];   // swizzled: (kv>>5)*2048 + bswz(kv&31, d)
  __shared__ float  Ss[64 * 64];
  __shared__ float  mrow[64];
  __shared__ float  lrow[64];
  __shared__ float  srow[64];
  __shared__ float  pred[256];

  const int tid  = threadIdx.x;
  const int w    = tid >> 5;
  const int lane = tid & 31;
  const int qt   = blockIdx.x;   // 0..15
  const int bh   = blockIdx.y;   // 0..127
  const int b    = bh >> 4;
  const int h    = bh & 15;
  const size_t rowstride = 3 * 1024;
  const size_t base = (size_t)b * 1024 * rowstride;
  const int hq = h * 64;

  // staging coords: 16 contiguous elems per thread per 64x64 tile
  const int sr = tid >> 2, sd = (tid & 3) * 16;

  {  // load Q tile (row-major)
    const __bf16* src = qkv + base + (size_t)(qt * 64 + sr) * rowstride + hq + sd;
    *(v8bf*)(Qs + sr * 64 + sd)     = *(const v8bf*)(src);
    *(v8bf*)(Qs + sr * 64 + sd + 8) = *(const v8bf*)(src + 8);
  }
  if (tid < 64) { mrow[tid] = -1e30f; lrow[tid] = 0.f; }

  const int mt = (w & 3) * 16;
  const int nb = (w >> 2) * 32;
  const int nt0 = nb >> 4;
  v8f o0 = {}; v8f o1 = {};
  const float scale = 0.07216878364870322f;  // 1/sqrt(3*C/H) = 1/sqrt(192)
  const int m_half = (lane & 16) ? 8 : 0;
  const int nlane  = lane & 15;

  for (int kt = 0; kt <= qt; ++kt) {
    const int kv0 = kt * 64;
    {  // load K row-major, V swizzled
      const size_t off = base + (size_t)(kv0 + sr) * rowstride + hq + sd;
      const __bf16* ksrc = qkv + off + 1024;
      *(v8bf*)(Ks + sr * 64 + sd)     = *(const v8bf*)(ksrc);
      *(v8bf*)(Ks + sr * 64 + sd + 8) = *(const v8bf*)(ksrc + 8);
      v8bf v0 = *(const v8bf*)(qkv + off + 2048);
      v8bf v1 = *(const v8bf*)(qkv + off + 2048 + 8);
      __bf16* vb = Vs + ((sr >> 5) << 11);
      const int kk = sr & 31;
#pragma unroll
      for (int j = 0; j < 8; ++j) {
        vb[bswz(kk, sd + j)]     = v0[j];
        vb[bswz(kk, sd + 8 + j)] = v1[j];
      }
    }
    __syncthreads();

    // S tile = Q K^T (reduction over D=64 -> 2 wmma per output tile)
    v8f s0 = {}; v8f s1 = {};
#pragma unroll
    for (int dc = 0; dc < 64; dc += 32) {
      v16bf a  = frag_rowK(Qs + mt * 64 + dc, 64);
      v16bf b0 = frag_rowK(Ks + nb * 64 + dc, 64);   // K^T: rows are S-columns
      v16bf b1 = frag_rowK(Ks + (nb + 16) * 64 + dc, 64);
      s0 = wmma_bf16(a, b0, s0);
      s1 = wmma_bf16(a, b1, s1);
    }
#pragma unroll
    for (int i = 0; i < 8; ++i) {   // scale + causal mask -> Ss
      int m = mt + i + m_half;
      int q = qt * 64 + m;
      int kvA = kv0 + nb + nlane;
      float vA = s0[i] * scale; if (kvA > q)      vA = -1e30f;
      float vB = s1[i] * scale; if (kvA + 16 > q) vB = -1e30f;
      Ss[m * 64 + nb + nlane]      = vA;
      Ss[m * 64 + nb + 16 + nlane] = vB;
    }
    __syncthreads();

    // online softmax: 4 threads per row
    {
      const int r = tid >> 2, qd = tid & 3;
      const float* srp = Ss + r * 64 + qd * 16;
      float mx = -1e30f;
#pragma unroll
      for (int j = 0; j < 16; ++j) mx = fmaxf(mx, srp[j]);
      pred[tid] = mx;
    }
    __syncthreads();
    if (tid < 64) {
      float mx = fmaxf(fmaxf(pred[tid * 4 + 0], pred[tid * 4 + 1]),
                       fmaxf(pred[tid * 4 + 2], pred[tid * 4 + 3]));
      mx = fmaxf(mx, mrow[tid]);
      srow[tid] = __expf(mrow[tid] - mx);
      mrow[tid] = mx;
    }
    __syncthreads();
    {
      const int r = tid >> 2, qd = tid & 3;
      const float mx = mrow[r];
      float* srp = Ss + r * 64 + qd * 16;
      float sum = 0.f;
#pragma unroll
      for (int j = 0; j < 16; ++j) {
        float pv = __expf(srp[j] - mx);
        srp[j] = pv;
        sum += pv;
      }
      pred[tid] = sum;
    }
    __syncthreads();
    if (tid < 64) {
      lrow[tid] = lrow[tid] * srow[tid] +
                  (pred[tid * 4 + 0] + pred[tid * 4 + 1] +
                   pred[tid * 4 + 2] + pred[tid * 4 + 3]);
    }
    __syncthreads();

    // rescale existing O, then O += P V (2 wmma per tile over kv chunks)
#pragma unroll
    for (int i = 0; i < 8; ++i) {
      float sc = srow[mt + i + m_half];
      o0[i] *= sc; o1[i] *= sc;
    }
#pragma unroll
    for (int kc = 0; kc < 64; kc += 32) {
      v16bf a  = frag_rowK_f32(Ss + mt * 64 + kc, 64);
      const __bf16* vb = Vs + ((kc >> 5) << 11);
      v16bf b0 = frag_swz(vb, nt0);
      v16bf b1 = frag_swz(vb, nt0 + 1);
      o0 = wmma_bf16(a, b0, o0);
      o1 = wmma_bf16(a, b1, o1);
    }
    __syncthreads();
  }

  // normalize and write y (bf16, [B,T,C] with head offset)
#pragma unroll
  for (int i = 0; i < 8; ++i) {
    int m = mt + i + m_half;
    float inv = 1.f / lrow[m];
    int t = qt * 64 + m;
    size_t o = ((size_t)b * 1024 + t) * 1024 + hq;
    y[o + nb + nlane]      = (__bf16)(o0[i] * inv);
    y[o + nb + 16 + nlane] = (__bf16)(o1[i] * inv);
  }
}

// ---------------------------------------------------------------------------
// Host-side orchestration
// ---------------------------------------------------------------------------
extern "C" void kernel_launch(void* const* d_in, const int* in_sizes, int n_in,
                              void* d_out, int out_size, void* d_ws, size_t ws_size,
                              hipStream_t stream) {
  const float* x     = (const float*)d_in[0];
  const float* Wqkv  = (const float*)d_in[1];
  const float* bqkv  = (const float*)d_in[2];
  const float* Wproj = (const float*)d_in[3];
  const float* bproj = (const float*)d_in[4];
  const float* ln1g  = (const float*)d_in[5];
  const float* ln1b  = (const float*)d_in[6];
  const float* ln2g  = (const float*)d_in[7];
  const float* ln2b  = (const float*)d_in[8];
  const float* W1    = (const float*)d_in[9];
  const float* b1    = (const float*)d_in[10];
  const float* W2    = (const float*)d_in[11];
  const float* b2    = (const float*)d_in[12];
  float* out = (float*)d_out;

  char* ws = (char*)d_ws;
  const size_t MB = 1ull << 20;
  __bf16* hln    = (__bf16*)(ws + 0);        // 8192x1024 bf16 (16 MB, reused)
  __bf16* qkv    = (__bf16*)(ws + 16 * MB);  // 8192x3072 bf16 (48 MB)
  __bf16* ybf    = (__bf16*)(ws + 64 * MB);  // 8192x1024 bf16 (16 MB)
  float*  x1     = (float*) (ws + 80 * MB);  // 8192x1024 f32  (32 MB)
  __bf16* hmlp   = (__bf16*)(ws + 112 * MB); // 8192x4096 bf16 (64 MB)
  __bf16* wqkvB  = (__bf16*)(ws + 176 * MB); // 6 MB
  __bf16* wprojB = (__bf16*)(ws + 182 * MB); // 2 MB
  __bf16* w1B    = (__bf16*)(ws + 184 * MB); // 8 MB
  __bf16* w2B    = (__bf16*)(ws + 192 * MB); // 8 MB

  const dim3 blk(256);

  f32_to_bf16<<<(3145728 + 255) / 256, blk, 0, stream>>>(Wqkv,  wqkvB,  3145728);
  f32_to_bf16<<<(1048576 + 255) / 256, blk, 0, stream>>>(Wproj, wprojB, 1048576);
  f32_to_bf16<<<(4194304 + 255) / 256, blk, 0, stream>>>(W1,    w1B,    4194304);
  f32_to_bf16<<<(4194304 + 255) / 256, blk, 0, stream>>>(W2,    w2B,    4194304);

  layernorm_bf16<<<8192, blk, 0, stream>>>(x, ln1g, ln1b, hln);
  // qkv = hln @ Wqkv + bqkv                      [8192,3072,1024]
  gemm_wmma<false, false><<<dim3(3072 / 64, 8192 / 128), blk, 0, stream>>>(
      hln, wqkvB, bqkv, nullptr, nullptr, qkv, 8192, 3072, 1024);
  flash_attn<<<dim3(16, 128), blk, 0, stream>>>(qkv, ybf);
  // x1 = x + y @ Wproj + bproj                   [8192,1024,1024]
  gemm_wmma<false, true><<<dim3(1024 / 64, 8192 / 128), blk, 0, stream>>>(
      ybf, wprojB, bproj, x, x1, nullptr, 8192, 1024, 1024);
  layernorm_bf16<<<8192, blk, 0, stream>>>(x1, ln2g, ln2b, hln);
  // hmlp = relu(hln @ W1 + b1)                   [8192,4096,1024]
  gemm_wmma<true, false><<<dim3(4096 / 64, 8192 / 128), blk, 0, stream>>>(
      hln, w1B, b1, nullptr, nullptr, hmlp, 8192, 4096, 1024);
  // out = x1 + hmlp @ W2 + b2                    [8192,1024,4096]
  gemm_wmma<false, true><<<dim3(1024 / 64, 8192 / 128), blk, 0, stream>>>(
      hmlp, w2B, b2, x1, out, nullptr, 8192, 1024, 4096);
}
